// ArcFaceLoss_22969485099312
// MI455X (gfx1250) — matrix operand order repaired and checked by reading, not compiled, not run on hardware
//
#include <hip/hip_runtime.h>

// ---------------- problem constants ----------------
#define N_      4096
#define D_      512
#define C_      30000
#define CHUNK   1024                    // classes per block (y-grid)
#define NCHUNK  30                      // ceil(30000/1024)
#define CPAD    (NCHUNK * CHUNK)        // 30720 padded class count
#define DP      520                     // LDS row stride (bf16 elems), bank-spread padding

#define SCALE_F   64.0f
#define MARGIN_F  0.5f
#define EPSC      1e-7f
#define EPSN      1e-12f
#define NEG_INF   (-__builtin_inff())
#define MAX_CLAMP (-3.0e38f)            // makes exp(-inf - msafe) == 0 branch-free

typedef __attribute__((ext_vector_type(4)))  float  v4f;
typedef __attribute__((ext_vector_type(8)))  float  v8f;
typedef __attribute__((ext_vector_type(4)))  __bf16 v4bf;
typedef __attribute__((ext_vector_type(8)))  __bf16 v8bf;
typedef __attribute__((ext_vector_type(16))) __bf16 v16bf;

// ---------------- kernel 1: L2-normalize classifier rows -> bf16 ----------------
// grid = CPAD blocks of 128 threads; pad rows (c >= C) zero-filled.
__global__ void normalize_weights_bf16(const float* __restrict__ w,
                                       __bf16* __restrict__ wn) {
  const int c = blockIdx.x;
  __bf16* orow = wn + (size_t)c * D_;
  const int d = threadIdx.x * 4;                 // 128 threads * 4 = 512 = D
  if (c >= C_) {
    v4bf z = {};
    *(v4bf*)(orow + d) = z;
    return;
  }
  const float* row = w + (size_t)c * D_;
  v4f v = *(const v4f*)(row + d);
  float ss = v[0]*v[0] + v[1]*v[1] + v[2]*v[2] + v[3]*v[3];
  #pragma unroll
  for (int off = 16; off > 0; off >>= 1) ss += __shfl_xor(ss, off, 32);
  __shared__ float wred[4];
  const int wv = threadIdx.x >> 5, ln = threadIdx.x & 31;
  if (ln == 0) wred[wv] = ss;
  __syncthreads();
  const float tot = wred[0] + wred[1] + wred[2] + wred[3];
  const float rn = 1.0f / fmaxf(sqrtf(tot), EPSN);
  v4bf o;
  #pragma unroll
  for (int i = 0; i < 4; ++i) o[i] = (__bf16)(v[i] * rn);
  *(v4bf*)(orow + d) = o;
}

// ---------------- kernel 2: embeddings fp32 -> bf16 ----------------
__global__ void convert_emb_bf16(const float* __restrict__ e, __bf16* __restrict__ eb) {
  const size_t i = ((size_t)blockIdx.x * blockDim.x + threadIdx.x) * 4;
  v4f v = *(const v4f*)(e + i);
  v4bf o;
  #pragma unroll
  for (int k = 0; k < 4; ++k) o[k] = (__bf16)v[k];
  *(v4bf*)(eb + i) = o;
}

// ---------------- kernel 3: fused WMMA GEMM + softmax partials ----------------
// grid.x = N/16 row tiles, grid.y = NCHUNK class chunks. 256 threads = 8 waves.
// Each wave owns 8 contiguous 16-column B tiles -> block covers 16 x 1024.
__global__ __launch_bounds__(256) void gemm_softmax_chunk(
    const __bf16* __restrict__ eb,    // [N, D] bf16
    const __bf16* __restrict__ wn,    // [CPAD, D] bf16, L2-normalized rows
    const int*    __restrict__ labels,
    float* __restrict__ part_max,     // [N, NCHUNK]
    float* __restrict__ part_sum,     // [N, NCHUNK]
    float* __restrict__ tlog)         // [N] logit at target class
{
  __shared__ __align__(16) __bf16 sA[16 * DP];
  __shared__ int   sLab[16];
  __shared__ float sRm[8 * 16];
  __shared__ float sRs[8 * 16];

  const int row_base   = blockIdx.x * 16;
  const int chunk      = blockIdx.y;
  const int chunk_base = chunk * CHUNK;
  const int tid        = threadIdx.x;

  // stage A tile (16 x 512 bf16) into LDS, 16B per thread per iter
  for (int q = tid; q < 16 * 64; q += 256) {
    const int r = q >> 6, off = (q & 63) * 8;
    *(v8bf*)(&sA[r * DP + off]) =
        *(const v8bf*)(eb + (size_t)(row_base + r) * D_ + off);
  }
  if (tid < 16) sLab[tid] = labels[row_base + tid];
  __syncthreads();

  const int wave = tid >> 5, lane = tid & 31;
  const int half = lane >> 4, lidx = lane & 15;

  v8f acc[8];
  #pragma unroll
  for (int t = 0; t < 8; ++t) { v8f z = {0,0,0,0,0,0,0,0}; acc[t] = z; }

  // A fragment source: lane's matrix row is lidx
  const __bf16* arow = &sA[lidx * DP];

  // Single per-lane B base; every load below is base + compile-time immediate.
  // wave's first column = chunk_base + wave*128; lane owns column (+lidx);
  // lanes 16-31 fetch the K=16..31 half (+half*16 elements).
  const int col0 = chunk_base + wave * 128 + lidx;
  const char* bbase = (const char*)(wn + (size_t)col0 * D_ + half * 16);

  // K loop: D=512 in 16 steps of 32 (fully unrolled -> immediate offsets).
  // All 8 B fragments of a K-step are staged into distinct registers BEFORE
  // the WMMAs so 16 b128 loads are in flight and waits count down partially
  // instead of the load->wait0->wmma serialization.
  #pragma unroll
  for (int ks = 0; ks < 16; ++ks) {
    const int kb = ks * 32;
    // 16-bit A 16x32 layout (contiguous, matches B): lanes 0-15 hold K[0..15]; lanes 16-31 K[16..31]
    v8bf alo = *(const v8bf*)(arow + kb + half * 16);
    v8bf ahi = *(const v8bf*)(arow + kb + half * 16 + 8);
    v16bf af = __builtin_shufflevector(alo, ahi,
        0,1,2,3,4,5,6,7,8,9,10,11,12,13,14,15);

    v16bf bf[8];
    #pragma unroll
    for (int t = 0; t < 8; ++t) {
      // tile stride 16 rows * 512 * 2B = 16384; K stride kb*2 -> 24-bit IOFFSET
      bf[t] = *(const v16bf*)(bbase + t * (16 * D_ * 2) + kb * 2);
    }
    #pragma unroll
    for (int t = 0; t < 8; ++t) {
      acc[t] = __builtin_amdgcn_wmma_f32_16x16x32_bf16(
          false, af, false, bf[t], (short)0, acc[t], false, false);
    }
  }

  // ---- epilogue pass 1: clip + margin + scale, logits kept in acc regs ----
  int lab[8];
  #pragma unroll
  for (int i = 0; i < 8; ++i) lab[i] = sLab[half * 8 + i];  // lanes>=16 hold rows 8..15

  #pragma unroll
  for (int t = 0; t < 8; ++t) {
    const int col = col0 + t * 16;
    const bool valid = col < C_;
    #pragma unroll
    for (int i = 0; i < 8; ++i) {
      float x = acc[t][i];
      x = fminf(fmaxf(x, -1.0f + EPSC), 1.0f - EPSC);
      float logit = SCALE_F * x;
      if (valid && col == lab[i]) {
        logit = SCALE_F * cosf(acosf(x) + MARGIN_F);
        tlog[row_base + half * 8 + i] = logit;   // unique writer per row
      }
      acc[t][i] = valid ? logit : NEG_INF;
    }
  }

  // ---- pass 2: per-row max (regs then 16-lane butterfly) ----
  float rmax[8];
  #pragma unroll
  for (int i = 0; i < 8; ++i) {
    float m = acc[0][i];
    #pragma unroll
    for (int t = 1; t < 8; ++t) m = fmaxf(m, acc[t][i]);
    rmax[i] = m;
  }
  #pragma unroll
  for (int off = 1; off < 16; off <<= 1) {
    #pragma unroll
    for (int i = 0; i < 8; ++i)
      rmax[i] = fmaxf(rmax[i], __shfl_xor(rmax[i], off, 32));
  }

  // ---- pass 3: one exp per element against shared max, add-butterfly ----
  float rsum[8];
  #pragma unroll
  for (int i = 0; i < 8; ++i) {
    const float msafe = fmaxf(rmax[i], MAX_CLAMP);
    float s = 0.0f;
    #pragma unroll
    for (int t = 0; t < 8; ++t) s += __expf(acc[t][i] - msafe);
    rsum[i] = s;
  }
  #pragma unroll
  for (int off = 1; off < 16; off <<= 1) {
    #pragma unroll
    for (int i = 0; i < 8; ++i)
      rsum[i] += __shfl_xor(rsum[i], off, 32);
  }

  if (lidx == 0) {
    #pragma unroll
    for (int i = 0; i < 8; ++i) {
      const int r = half * 8 + i;
      sRm[wave * 16 + r] = rmax[i];
      sRs[wave * 16 + r] = rsum[i];
    }
  }
  __syncthreads();

  // cross-wave merge: threads 0..15 each own one row
  if (tid < 16) {
    float M = NEG_INF;
    #pragma unroll
    for (int wv = 0; wv < 8; ++wv) M = fmaxf(M, sRm[wv * 16 + tid]);
    const float msafe = fmaxf(M, MAX_CLAMP);
    float S = 0.0f;
    #pragma unroll
    for (int wv = 0; wv < 8; ++wv)
      S += sRs[wv * 16 + tid] * __expf(sRm[wv * 16 + tid] - msafe);
    part_max[(size_t)(row_base + tid) * NCHUNK + chunk] = M;
    part_sum[(size_t)(row_base + tid) * NCHUNK + chunk] = S;
  }
}

// ---------------- kernel 4: merge chunk partials -> per-row NLL ----------------
__global__ void finalize_rows(const float* __restrict__ pm,
                              const float* __restrict__ ps,
                              const float* __restrict__ tl,
                              float* __restrict__ nll) {
  const int r = blockIdx.x * blockDim.x + threadIdx.x;
  if (r >= N_) return;
  float M = NEG_INF;
  #pragma unroll
  for (int j = 0; j < NCHUNK; ++j) M = fmaxf(M, pm[(size_t)r * NCHUNK + j]);
  float S = 0.0f;
  #pragma unroll
  for (int j = 0; j < NCHUNK; ++j)
    S += ps[(size_t)r * NCHUNK + j] * __expf(pm[(size_t)r * NCHUNK + j] - M);
  nll[r] = M + logf(S) - tl[r];
}

// ---------------- kernel 5: deterministic mean ----------------
__global__ void reduce_mean(const float* __restrict__ nll, float* __restrict__ out) {
  __shared__ float red[256];
  float s = 0.0f;
  for (int r = threadIdx.x; r < N_; r += 256) s += nll[r];
  red[threadIdx.x] = s;
  __syncthreads();
  for (int st = 128; st > 0; st >>= 1) {
    if (threadIdx.x < st) red[threadIdx.x] += red[threadIdx.x + st];
    __syncthreads();
  }
  if (threadIdx.x == 0) out[0] = red[0] / (float)N_;
}

// ---------------- launch ----------------
extern "C" void kernel_launch(void* const* d_in, const int* in_sizes, int n_in,
                              void* d_out, int out_size, void* d_ws, size_t ws_size,
                              hipStream_t stream) {
  const float* emb    = (const float*)d_in[0];   // [N, D]
  const int*   labels = (const int*)d_in[1];     // [N]
  const float* w      = (const float*)d_in[2];   // [C, D]

  char* ws = (char*)d_ws;
  __bf16* wn = (__bf16*)ws;                 ws += (size_t)CPAD * D_ * sizeof(__bf16);
  __bf16* eb = (__bf16*)ws;                 ws += (size_t)N_   * D_ * sizeof(__bf16);
  float* pm  = (float*)ws;                  ws += (size_t)N_ * NCHUNK * sizeof(float);
  float* psum= (float*)ws;                  ws += (size_t)N_ * NCHUNK * sizeof(float);
  float* tl  = (float*)ws;                  ws += (size_t)N_ * sizeof(float);
  float* nll = (float*)ws;

  normalize_weights_bf16<<<CPAD, 128, 0, stream>>>(w, wn);
  convert_emb_bf16<<<(N_ * D_ / 4) / 256, 256, 0, stream>>>(emb, eb);

  dim3 grid(N_ / 16, NCHUNK);
  gemm_softmax_chunk<<<grid, 256, 0, stream>>>(eb, wn, labels, pm, psum, tl);

  finalize_rows<<<N_ / 256, 256, 0, stream>>>(pm, psum, tl, nll);
  reduce_mean<<<1, 256, 0, stream>>>(nll, (float*)d_out);
}
